// theModel_14843406975629
// MI455X (gfx1250) — matrix operand order repaired and checked by reading, not compiled, not run on hardware
//
#include <hip/hip_runtime.h>

// ---------------------------------------------------------------------------
// MI455X (gfx1250) implementation: all matmuls on v_wmma_f32_16x16x32_f16,
// shared operand tiles moved via GLOBAL_LOAD_ASYNC_TO_LDS_B128 (ASYNCcnt),
// double-buffered so async copy overlaps WMMA.
//   h = (embed||bits) @ Wc + bc
//   4x { h += bandedAttn(rms(h)); h += (silu(z@w1)*(z@w3)) @ w2 }  z=rms(h)
//   out = rms(h) @ head_w + head_b
// ---------------------------------------------------------------------------

typedef _Float16 half_t;
typedef __attribute__((ext_vector_type(16))) _Float16 v16h;
typedef __attribute__((ext_vector_type(8)))  float    v8f;

#define DEVFN static __device__ __forceinline__

DEVFN v8f wmma16(v16h a, v16h b, v8f c) {
  // (neg_a, A, neg_b, B, c_mod, C, reuse_a, reuse_b)
  return __builtin_amdgcn_wmma_f32_16x16x32_f16(false, a, false, b, (short)0, c,
                                                false, false);
}

union FragU { uint4 q[2]; v16h v; };

// A-matrix 16x32 f16 fragment (row = lane&15). Per ISA: element e maps to
// k = ((e>>3)<<4) + half*8 + (e&7)  => two contiguous 16B runs per lane.
DEVFN v16h load_a_frag(const half_t* tile, int ld) {
  int lane = threadIdx.x & 31;
  int r = lane & 15, hs = lane >> 4;
  const half_t* p = tile + r * ld + hs * 8;
  FragU u;
  u.q[0] = *reinterpret_cast<const uint4*>(p);
  u.q[1] = *reinterpret_cast<const uint4*>(p + 16);
  return u.v;
}

// B-matrix 32x16 f16 fragment read from BT[N][K] (col = lane&15). Same k map
// as A: element e -> k = ((e>>3)<<4) + half*8 + (e&7) => two 16B runs.
DEVFN v16h load_b_frag(const half_t* bt, int ld) {
  int lane = threadIdx.x & 31;
  int n = lane & 15, hs = lane >> 4;
  const half_t* p = bt + n * ld + hs * 8;
  FragU u;
  u.q[0] = *reinterpret_cast<const uint4*>(p);
  u.q[1] = *reinterpret_cast<const uint4*>(p + 16);
  return u.v;
}

// Async 16B global -> LDS copy (per-lane addresses, ASYNCcnt-tracked).
DEVFN void async_copy16(unsigned lds_byte_addr, const void* gptr) {
  asm volatile("global_load_async_to_lds_b128 %0, %1, off"
               :: "v"(lds_byte_addr),
                  "v"((unsigned long long)(size_t)gptr)
               : "memory");
}
DEVFN void wait_async() {
#if __has_builtin(__builtin_amdgcn_s_wait_asynccnt)
  __builtin_amdgcn_s_wait_asynccnt(0);
#else
  asm volatile("s_wait_asynccnt 0" ::: "memory");
#endif
}

// ---------------------------------------------------------------------------
// Generic WMMA GEMM. Block = 256 thr (8 waves); wave tile 32x64 (2 A-frags x
// 4 B-frags = 8 wmma / K-step); block tile 256x64. The 64x32 B tile is shared
// by all waves: async-staged to LDS, double buffered. Epilogue by MODE:
//   0: f32 store (col<N guarded)                      [input proj, head]
//   1: f32 residual add                               [FFN w2]
//   2: f16 store in [B,H,S,Dh] layout, * oscale       [QKV]
//   3: dual GEMM, f16 store silu(C1+b1)*(C2+b3)       [FFN w1/w3 fused]
// ---------------------------------------------------------------------------
template<int MODE>
__global__ __launch_bounds__(256) void gemm16(
    const half_t* __restrict__ A, int lda,
    const half_t* __restrict__ BT,
    const half_t* __restrict__ BT2,
    int kdim,
    const float* __restrict__ bias,
    const float* __restrict__ bias2,
    float* __restrict__ of32,
    half_t* __restrict__ of16,
    int N, int ldo, float oscale)
{
  __shared__ __align__(16) half_t Bs[2 * 64 * 32];
  __shared__ __align__(16) half_t Bs2[(MODE == 3) ? 2 * 64 * 32 : 8];

  const int lane = threadIdx.x & 31;
  const int wv   = threadIdx.x >> 5;
  const int m0   = blockIdx.x * 256 + wv * 32;
  const int n0   = blockIdx.y * 64;
  const int hs   = lane >> 4, cl = lane & 15;

  const unsigned bs0 = (unsigned)(size_t)&Bs[0];
  const unsigned bs1 = (unsigned)(size_t)&Bs2[0];
  const int tn = threadIdx.x >> 2;   // 0..63: B row
  const int tc = threadIdx.x & 3;    // 16B chunk within 32-half row

  const v8f zero8 = {0.f,0.f,0.f,0.f,0.f,0.f,0.f,0.f};
  v8f acc[2][4], acc2[2][4];
#pragma unroll
  for (int r2 = 0; r2 < 2; ++r2)
#pragma unroll
    for (int j = 0; j < 4; ++j) { acc[r2][j] = zero8; acc2[r2][j] = zero8; }

  // stage B tile for K-step kk into buffer buf (one 16B async op per thread)
  auto stage = [&](int buf, int kk) {
    const half_t* g = BT + (size_t)(n0 + tn) * kdim + kk + tc * 8;
    async_copy16(bs0 + (unsigned)(buf * 2048 + tn * 32 + tc * 8) * 2, g);
    if (MODE == 3) {
      const half_t* g2 = BT2 + (size_t)(n0 + tn) * kdim + kk + tc * 8;
      async_copy16(bs1 + (unsigned)(buf * 2048 + tn * 32 + tc * 8) * 2, g2);
    }
  };

  stage(0, 0);
  int buf = 0;
  for (int k = 0; k < kdim; k += 32, buf ^= 1) {
    wait_async();
    __syncthreads();                       // Bs[buf] visible to all waves
    if (k + 32 < kdim) stage(buf ^ 1, k + 32);
    if (k + 64 < kdim)                     // prefetch A stream
      __builtin_prefetch(A + (size_t)m0 * lda + k + 64, 0, 0);
    v16h a0 = load_a_frag(A + (size_t)m0 * lda + k, lda);
    v16h a1 = load_a_frag(A + (size_t)(m0 + 16) * lda + k, lda);
#pragma unroll
    for (int j = 0; j < 4; ++j) {
      v16h b = load_b_frag(&Bs[buf * 2048 + j * 16 * 32], 32);
      acc[0][j] = wmma16(a0, b, acc[0][j]);
      acc[1][j] = wmma16(a1, b, acc[1][j]);
      if (MODE == 3) {
        v16h b2 = load_b_frag(&Bs2[buf * 2048 + j * 16 * 32], 32);
        acc2[0][j] = wmma16(a0, b2, acc2[0][j]);
        acc2[1][j] = wmma16(a1, b2, acc2[1][j]);
      }
    }
    __syncthreads();                       // reads done before restage
  }

#pragma unroll
  for (int r2 = 0; r2 < 2; ++r2) {
#pragma unroll
    for (int j = 0; j < 4; ++j) {
      int col = n0 + j * 16 + cl;
      float bv  = (col < N) ? bias[col] : 0.f;
      float bv2 = 0.f;
      if (MODE == 3) bv2 = (col < N) ? bias2[col] : 0.f;
#pragma unroll
      for (int i = 0; i < 8; ++i) {
        int m = m0 + r2 * 16 + i + 8 * hs;   // C/D layout: row = i + 8*half
        float val = acc[r2][j][i] + bv;
        if (MODE == 0) {
          if (col < N) of32[(size_t)m * ldo + col] = val;
        } else if (MODE == 1) {
          of32[(size_t)m * ldo + col] += val;
        } else if (MODE == 2) {
          int hh = col >> 6, dh = col & 63;
          int bb = m >> 13, s = m & 8191;
          of16[((((size_t)bb * 8) + hh) * 8192 + s) * 64 + dh] =
              (half_t)(val * oscale);
        } else {
          float u3 = acc2[r2][j][i] + bv2;
          float sv = val / (1.f + __expf(-val));   // silu
          of16[(size_t)m * ldo + col] = (half_t)(sv * u3);
        }
      }
    }
  }
}

// ---------------------------------------------------------------------------
// RMSNorm: wave per row of 512, f32 in -> f16 out (GEMM A operand).
// ---------------------------------------------------------------------------
__global__ __launch_bounds__(256) void rmsnorm_k(
    const float* __restrict__ x, const float* __restrict__ w,
    half_t* __restrict__ out)
{
  int lane = threadIdx.x & 31, wv = threadIdx.x >> 5;
  int row = blockIdx.x * 8 + wv;
  const float* xr = x + (size_t)row * 512;
  float ss = 0.f;
#pragma unroll
  for (int c = 0; c < 16; ++c) { float v = xr[lane + c * 32]; ss += v * v; }
#pragma unroll
  for (int off = 16; off > 0; off >>= 1) ss += __shfl_xor(ss, off, 32);
  float r = rsqrtf(ss * (1.f / 512.f) + 1e-6f);
#pragma unroll
  for (int c = 0; c < 16; ++c) {
    int cc = lane + c * 32;
    out[(size_t)row * 512 + cc] = (half_t)(xr[cc] * r * w[cc]);
  }
}

// ---------------------------------------------------------------------------
// Embedding + MSB-first binary id features -> X0 [16384][512] f16.
// ---------------------------------------------------------------------------
__global__ void embed_k(const int* __restrict__ x, const int* __restrict__ ids,
                        const float* __restrict__ table, half_t* __restrict__ X0)
{
  int idx = blockIdx.x * 256 + threadIdx.x;    // exact grid, no guard
  int m = idx >> 9, c = idx & 511;
  float v;
  if (c < 480) v = table[x[m] * 480 + c] * 21.908902300206645f; // sqrt(480)
  else         v = (float)((ids[m] >> (31 - (c - 480))) & 1);
  X0[idx] = (half_t)v;
}

// ---------------------------------------------------------------------------
// Weight pack: f32 W[K][Nn] -> f16 WT[Np][K] (transposed, zero padded).
// ---------------------------------------------------------------------------
__global__ void packT_k(const float* __restrict__ W, half_t* __restrict__ WT,
                        int K, int Nn, int Np)
{
  int idx = blockIdx.x * 256 + threadIdx.x;
  if (idx >= Np * K) return;
  int n = idx / K, kk = idx - n * K;
  float v = (n < Nn) ? W[(size_t)kk * Nn + n] : 0.f;
  WT[idx] = (half_t)v;
}

// ---------------------------------------------------------------------------
// Banded attention, one block per (batch, head, query-chunk of 128).
// Flash-style over 3 key blocks of 128. K block async-staged to LDS
// (EXEC-masked OOB; garbage only reaches band-masked score elements),
// V staged transposed + zero-filled, P re-fragmented through LDS (aliases
// the K buffer; barrier-separated). h += O.
// ---------------------------------------------------------------------------
__global__ __launch_bounds__(256) void attn_k(
    const half_t* __restrict__ qg, const half_t* __restrict__ kg,
    const half_t* __restrict__ vg, float* __restrict__ hio)
{
  const int c  = blockIdx.x;   // chunk 0..63
  const int hh = blockIdx.y;   // head 0..7
  const int bb = blockIdx.z;   // batch 0..1
  const int S = 8192;
  const half_t* qb  = qg + (size_t)((bb * 8 + hh) * S) * 64;
  const half_t* kb_ = kg + (size_t)((bb * 8 + hh) * S) * 64;
  const half_t* vb_ = vg + (size_t)((bb * 8 + hh) * S) * 64;

  __shared__ __align__(16) half_t smem[16384 + 64 * 136];
  half_t* Pb  = smem;            // [8][16][128] probs (aliases Ks)
  half_t* Ks  = smem;            // [128][64] K block
  half_t* VTb = smem + 16384;    // [64][136] V^T block (+pad)

  const int lane = threadIdx.x & 31;
  const int wv   = threadIdx.x >> 5;
  const int hs   = lane >> 4, cl = lane & 15;
  const int r0   = wv * 16;
  const int q0   = c * 128;
  half_t* Pw = Pb + wv * 2048;
  const unsigned ksOff = (unsigned)(size_t)Ks;

  // Q fragments (K-dim = Dh = 64), q pre-scaled by 1/8 at pack time.
  v16h qa0 = load_a_frag(qb + (size_t)(q0 + r0) * 64, 64);
  v16h qa1 = load_a_frag(qb + (size_t)(q0 + r0) * 64 + 32, 64);

  const v8f zero8 = {0.f,0.f,0.f,0.f,0.f,0.f,0.f,0.f};
  float mrow[8], lrow[8];
  v8f O[4];
#pragma unroll
  for (int i = 0; i < 8; ++i) { mrow[i] = -1e30f; lrow[i] = 0.f; }
#pragma unroll
  for (int j = 0; j < 4; ++j) O[j] = zero8;

  for (int kbi = 0; kbi < 3; ++kbi) {
    int key0 = (c - 1) * 128 + kbi * 128;   // may be <0 or >=S (masked)
    __syncthreads();                        // prior P / VT / Ks reads done
    // Async-stage K block [128][64] (valid rows only).
    for (int t = threadIdx.x; t < 1024; t += 256) {
      int key = t >> 3, c8 = t & 7;
      int kpos = key0 + key;
      if (kpos >= 0 && kpos < S)
        async_copy16(ksOff + (unsigned)(key * 64 + c8 * 8) * 2,
                     kb_ + (size_t)kpos * 64 + c8 * 8);
    }
    // Stage V^T (zero outside sequence => P*V NaN-safe).
    for (int t = threadIdx.x; t < 128 * 64; t += 256) {
      int key = t >> 6, d = t & 63;
      int kpos = key0 + key;
      half_t val = (half_t)0.f;
      if (kpos >= 0 && kpos < S) val = vb_[(size_t)kpos * 64 + d];
      VTb[d * 136 + key] = val;
    }
    wait_async();
    __syncthreads();

    // scores 16x128: Q(16x64) x K^T(64x128), K from LDS
    v8f sc[8];
#pragma unroll
    for (int j = 0; j < 8; ++j) sc[j] = zero8;
#pragma unroll
    for (int j = 0; j < 8; ++j) {
      v16h b0 = load_b_frag(Ks + (j * 16) * 64, 64);
      v16h b1 = load_b_frag(Ks + (j * 16) * 64 + 32, 64);
      sc[j] = wmma16(qa0, b0, sc[j]);
      sc[j] = wmma16(qa1, b1, sc[j]);
    }
    __syncthreads();     // all waves done reading Ks before P overwrites it

    // Online softmax per row (row = i + 8*half, col = lane&15 + j*16).
#pragma unroll
    for (int i = 0; i < 8; ++i) {
      int row = i + 8 * hs;
      int qpos = q0 + r0 + row;
      float lm = -1e30f;
#pragma unroll
      for (int j = 0; j < 8; ++j) {
        int jpos = key0 + j * 16 + cl;
        int rel = jpos - qpos;
        bool ok = (jpos >= 0) && (jpos < S) && (rel >= -128) && (rel <= 128);
        float val = ok ? sc[j][i] : -1e30f;
        sc[j][i] = val;
        lm = fmaxf(lm, val);
      }
#pragma unroll
      for (int off = 1; off < 16; off <<= 1)
        lm = fmaxf(lm, __shfl_xor(lm, off, 32));
      float mo = mrow[i];
      float mn = fmaxf(mo, lm);
      float scale = __expf(mo - mn);       // ->0 when mo = -inf
      float rs = 0.f;
#pragma unroll
      for (int j = 0; j < 8; ++j) {
        float v0 = sc[j][i];
        float p = (v0 <= -1e29f) ? 0.f : __expf(v0 - mn);
        sc[j][i] = p;
        rs += p;
      }
#pragma unroll
      for (int off = 1; off < 16; off <<= 1) rs += __shfl_xor(rs, off, 32);
      lrow[i] = lrow[i] * scale + rs;
      mrow[i] = mn;
#pragma unroll
      for (int j = 0; j < 4; ++j) O[j][i] *= scale;
    }

    // Re-fragment P through LDS (C-layout -> A-layout).
#pragma unroll
    for (int j = 0; j < 8; ++j)
#pragma unroll
      for (int i = 0; i < 8; ++i)
        Pw[(i + 8 * hs) * 128 + j * 16 + cl] = (half_t)sc[j][i];
    __syncthreads();

    // O += P(16x128) x V(128x64)
    for (int kk = 0; kk < 128; kk += 32) {
      v16h pa = load_a_frag(Pw + kk, 128);
#pragma unroll
      for (int j2 = 0; j2 < 4; ++j2) {
        v16h vf = load_b_frag(VTb + (j2 * 16) * 136 + kk, 136);
        O[j2] = wmma16(pa, vf, O[j2]);
      }
    }
  }

  // h += O / l   (each element owned by exactly one block)
#pragma unroll
  for (int j2 = 0; j2 < 4; ++j2) {
    int dh = j2 * 16 + cl;
#pragma unroll
    for (int i = 0; i < 8; ++i) {
      int row = i + 8 * hs;
      int qpos = q0 + r0 + row;
      float val = O[j2][i] / lrow[i];
      hio[((size_t)bb * S + qpos) * 512 + hh * 64 + dh] += val;
    }
  }
}

// ---------------------------------------------------------------------------
// Host orchestration
// ---------------------------------------------------------------------------
extern "C" void kernel_launch(void* const* d_in, const int* in_sizes, int n_in,
                              void* d_out, int out_size, void* d_ws, size_t ws_size,
                              hipStream_t stream) {
  (void)in_sizes; (void)n_in; (void)out_size; (void)ws_size;
  const int*   x           = (const int*)d_in[0];
  const int*   ids         = (const int*)d_in[1];
  const float* embed_table = (const float*)d_in[2];
  const float* Wc  = (const float*)d_in[3];
  const float* bc  = (const float*)d_in[4];
  const float* Wq  = (const float*)d_in[5];
  const float* bq  = (const float*)d_in[6];
  const float* Wk  = (const float*)d_in[7];
  const float* bk  = (const float*)d_in[8];
  const float* Wv  = (const float*)d_in[9];
  const float* bv  = (const float*)d_in[10];
  const float* w1  = (const float*)d_in[11];
  const float* b1  = (const float*)d_in[12];
  const float* w2  = (const float*)d_in[13];
  const float* b2  = (const float*)d_in[14];
  const float* w3  = (const float*)d_in[15];
  const float* b3  = (const float*)d_in[16];
  const float* an_w   = (const float*)d_in[17];
  const float* fn_w   = (const float*)d_in[18];
  const float* norm_w = (const float*)d_in[19];
  const float* head_w = (const float*)d_in[20];
  const float* head_b = (const float*)d_in[21];

  char* ws = (char*)d_ws;
  size_t off = 0;
  float*  hbuf = (float*)(ws + off);  off += (size_t)16384 * 512 * 4;   // h f32
  half_t* a16  = (half_t*)(ws + off); off += (size_t)16384 * 512 * 2;   // X0 / rms out
  half_t* r1   = (half_t*)(ws + off); off += (size_t)16384 * 1536 * 2;  // qkv | ffn-u
  half_t* wt   = (half_t*)(ws + off); off += (size_t)12877824 * 2;      // f16 weights

  half_t* qbuf = r1;
  half_t* kbuf = r1 + (size_t)16384 * 512;
  half_t* vbuf = r1 + (size_t)2 * 16384 * 512;
  half_t* ubuf = r1;                       // aliases qkv (disjoint phases)

  half_t* WcT   = wt;                                // [512][512]
  half_t* headT = wt + 262144 + (size_t)4 * 3145728; // [64][512] padded

  auto layerWT = [&](int i) { return wt + 262144 + (size_t)i * 3145728; };
  auto pack = [&](const float* W, half_t* WT, int K, int Nn, int Np) {
    int total = Np * K;
    packT_k<<<(total + 255) / 256, 256, 0, stream>>>(W, WT, K, Nn, Np);
  };

  // Pack all weights transposed f32 -> f16.
  pack(Wc, WcT, 512, 512, 512);
  for (int i = 0; i < 4; ++i) {
    half_t* lw = layerWT(i);
    pack(Wq + (size_t)i * 512 * 512,  lw + 0,       512, 512, 512);
    pack(Wk + (size_t)i * 512 * 512,  lw + 262144,  512, 512, 512);
    pack(Wv + (size_t)i * 512 * 512,  lw + 524288,  512, 512, 512);
    pack(w1 + (size_t)i * 512 * 1536, lw + 786432,  512, 1536, 1536);
    pack(w3 + (size_t)i * 512 * 1536, lw + 1572864, 512, 1536, 1536);
    pack(w2 + (size_t)i * 1536 * 512, lw + 2359296, 1536, 512, 512);
  }
  pack(head_w, headT, 512, 26, 64);

  // X0 = [embed*sqrt(480) || bits(ids)]
  embed_k<<<16384 * 512 / 256, 256, 0, stream>>>(x, ids, embed_table, a16);

  // h = X0 @ Wc + bc
  gemm16<0><<<dim3(64, 8), 256, 0, stream>>>(a16, 512, WcT, nullptr, 512,
      bc, nullptr, hbuf, nullptr, 512, 512, 1.f);

  for (int i = 0; i < 4; ++i) {
    half_t* lw = layerWT(i);
    // z = rms(h, an_w); q/k/v = z@W + b  (q scaled by 1/sqrt(64))
    rmsnorm_k<<<2048, 256, 0, stream>>>(hbuf, an_w + i * 512, a16);
    gemm16<2><<<dim3(64, 8), 256, 0, stream>>>(a16, 512, lw + 0, nullptr, 512,
        bq + i * 512, nullptr, nullptr, qbuf, 512, 0, 0.125f);
    gemm16<2><<<dim3(64, 8), 256, 0, stream>>>(a16, 512, lw + 262144, nullptr, 512,
        bk + i * 512, nullptr, nullptr, kbuf, 512, 0, 1.f);
    gemm16<2><<<dim3(64, 8), 256, 0, stream>>>(a16, 512, lw + 524288, nullptr, 512,
        bv + i * 512, nullptr, nullptr, vbuf, 512, 0, 1.f);
    // h += banded attention
    attn_k<<<dim3(64, 8, 2), 256, 0, stream>>>(qbuf, kbuf, vbuf, hbuf);
    // z = rms(h, fn_w); u = silu(z@w1+b1)*(z@w3+b3); h += u@w2 + b2
    rmsnorm_k<<<2048, 256, 0, stream>>>(hbuf, fn_w + i * 512, a16);
    gemm16<3><<<dim3(64, 24), 256, 0, stream>>>(a16, 512, lw + 786432,
        lw + 1572864, 512, b1 + i * 1536, b3 + i * 1536, nullptr, ubuf,
        1536, 1536, 1.f);
    gemm16<1><<<dim3(64, 8), 256, 0, stream>>>(ubuf, 1536, lw + 2359296,
        nullptr, 1536, b2 + i * 512, nullptr, hbuf, nullptr, 512, 512, 1.f);
  }

  // out = rms(h, norm_w) @ head_w + head_b
  rmsnorm_k<<<2048, 256, 0, stream>>>(hbuf, norm_w, a16);
  gemm16<0><<<dim3(64, 1), 256, 0, stream>>>(a16, 512, headT, nullptr, 512,
      head_b, nullptr, (float*)d_out, nullptr, 26, 26, 1.f);
}
